// napl_conv_withSAT_9302899163164
// MI455X (gfx1250) — matrix operand order repaired and checked by reading, not compiled, not run on hardware
//
#include <hip/hip_runtime.h>
#include <hip/hip_bf16.h>

typedef __attribute__((ext_vector_type(16))) _Float16 v16h;
typedef __attribute__((ext_vector_type(8)))  _Float16 v8h;
typedef __attribute__((ext_vector_type(8)))  float    v8f;

#define N_NODES 2048
#define B_SZ 8
#define F_IN 16
#define T_SZ 12
#define F_OUT 32
#define COLS (F_OUT * T_SZ)   // 384, col = o*12 + t

union FragA { v16h v; v8h h[2]; };

// ---------------------------------------------------------------------------
// Stage 1: supports = softmax(relu(E E^T), axis=1), stored f16 row-major (S)
// and transposed (St). One workgroup (256 thr) per row n.
// ---------------------------------------------------------------------------
__global__ void supports_kernel(const float* __restrict__ E,
                                _Float16* __restrict__ S,
                                _Float16* __restrict__ St) {
  const int n = blockIdx.x;
  const int tid = threadIdx.x;
  __shared__ float en[16];
  __shared__ float red[256];
  if (tid < 16) en[tid] = E[n * 16 + tid];
  __syncthreads();

  float v[8];
  float mx = -3.0e38f;
  #pragma unroll
  for (int i = 0; i < 8; ++i) {
    const int m = tid + i * 256;
    const float* em = E + m * 16;
    float d = 0.0f;
    #pragma unroll
    for (int f = 0; f < 16; ++f) d += en[f] * em[f];
    d = fmaxf(d, 0.0f);           // relu
    v[i] = d;
    mx = fmaxf(mx, d);
  }
  red[tid] = mx; __syncthreads();
  for (int s = 128; s > 0; s >>= 1) {
    if (tid < s) red[tid] = fmaxf(red[tid], red[tid + s]);
    __syncthreads();
  }
  mx = red[0]; __syncthreads();

  float sum = 0.0f;
  #pragma unroll
  for (int i = 0; i < 8; ++i) { v[i] = __expf(v[i] - mx); sum += v[i]; }
  red[tid] = sum; __syncthreads();
  for (int s = 128; s > 0; s >>= 1) {
    if (tid < s) red[tid] += red[tid + s];
    __syncthreads();
  }
  const float inv = 1.0f / red[0];

  #pragma unroll
  for (int i = 0; i < 8; ++i) {
    const int m = tid + i * 256;
    const _Float16 h = (_Float16)(v[i] * inv);
    S [(size_t)n * N_NODES + m] = h;
    St[(size_t)m * N_NODES + n] = h;
  }
}

// ---------------------------------------------------------------------------
// Stage 2: T2 = 2*S@S - I  (f16 out, fp32 WMMA accumulate).
// Block = 256 thr (8 waves): covers 128 rows x 64 cols. Grid (16, 32).
// A frags from S (row-major, l-contig), B frags from St (col j, l-contig).
// ---------------------------------------------------------------------------
__global__ void __launch_bounds__(256, 2)
cheb2_kernel(const _Float16* __restrict__ S,
             const _Float16* __restrict__ St,
             _Float16* __restrict__ T2) {
  const int tid = threadIdx.x;
  const int wave = tid >> 5, lane = tid & 31;
  const int lm = lane & 15, lhi = lane >> 4;
  const int i0 = blockIdx.x * 128 + wave * 16;
  const int j0 = blockIdx.y * 64;

  v8f acc[4] = {};
  const _Float16* arow = S + (size_t)(i0 + lm) * N_NODES + lhi * 8;

  for (int l = 0; l < N_NODES; l += 32) {
    FragA A;
    A.h[0] = *(const v8h*)(arow + l);        // K = l+{0..7} / {8..15}
    A.h[1] = *(const v8h*)(arow + l + 16);   // K = l+{16..23} / {24..31}
    #pragma unroll
    for (int ct = 0; ct < 4; ++ct) {
      const _Float16* bcol = St + (size_t)(j0 + ct * 16 + lm) * N_NODES + l + lhi * 8;
      FragA Bf;
      Bf.h[0] = *(const v8h*)(bcol);
      Bf.h[1] = *(const v8h*)(bcol + 16);
      acc[ct] = __builtin_amdgcn_wmma_f32_16x16x32_f16(
          false, A.v, false, Bf.v, (short)0, acc[ct], false, false);
    }
  }
  #pragma unroll
  for (int ct = 0; ct < 4; ++ct) {
    #pragma unroll
    for (int r = 0; r < 8; ++r) {
      const int i = i0 + r + lhi * 8;
      const int j = j0 + ct * 16 + lm;
      const float d = 2.0f * acc[ct][r] - (i == j ? 1.0f : 0.0f);
      T2[(size_t)i * N_NODES + j] = (_Float16)d;
    }
  }
}

// ---------------------------------------------------------------------------
// Stage 3: y[k][b][col][n] = sum_f x[b,n,f,t] * theta[k,f,o], col = o*12+t,
// stored f16 with n contiguous (feeds WMMA B frags directly).
// Block per (n-chunk of 64, b). Grid (32, 8).
// ---------------------------------------------------------------------------
__global__ void y_kernel(const float* __restrict__ x,
                         const float* __restrict__ theta,
                         _Float16* __restrict__ Y) {
  const int n0 = blockIdx.x * 64;
  const int b = blockIdx.y;
  const int tid = threadIdx.x;
  __shared__ float xs[64 * 192];
  __shared__ float th[3 * 16 * 32];

  const float* xb = x + ((size_t)b * N_NODES + n0) * 192;
  for (int i = tid; i < 64 * 192; i += 256) xs[i] = xb[i];
  for (int i = tid; i < 1536; i += 256) th[i] = theta[i];
  __syncthreads();

  for (int trip = tid; trip < 3 * F_OUT * T_SZ; trip += 256) {
    const int t = trip % T_SZ;
    const int o = (trip / T_SZ) % F_OUT;
    const int k = trip / (T_SZ * F_OUT);
    float w[16];
    #pragma unroll
    for (int f = 0; f < 16; ++f) w[f] = th[(k * 16 + f) * 32 + o];
    _Float16* yp = Y + ((size_t)(k * B_SZ + b) * COLS + o * T_SZ + t) * N_NODES + n0;
    for (int n = 0; n < 64; ++n) {
      float s = 0.0f;
      #pragma unroll
      for (int f = 0; f < 16; ++f) s += w[f] * xs[n * 192 + f * 12 + t];
      yp[n] = (_Float16)s;
    }
  }
}

// ---------------------------------------------------------------------------
// Stage 4: out[b,m,col] = relu( attn[b,m,m]*y0[b,m,col]
//            + sum_{k=1,2} sum_n (attn[b,n,m]*P_k[n,m]) * y_k[b,n,col] )
// Block = 256 thr (8 waves): m-tile of 64, all 384 cols. Grid (32, 8).
// A-frags staged via LDS (attn is [n][m]-major); B-frags from Y (n-contig).
// ---------------------------------------------------------------------------
__global__ void __launch_bounds__(256, 1)
agcn_main_kernel(const float* __restrict__ attn,
                 const _Float16* __restrict__ P1,
                 const _Float16* __restrict__ P2,
                 const _Float16* __restrict__ Y,
                 float* __restrict__ out) {
  const int m0 = blockIdx.x * 64;
  const int b = blockIdx.y;
  const int tid = threadIdx.x;
  const int wave = tid >> 5, lane = tid & 31;
  const int lm = lane & 15, lhi = lane >> 4;

  __shared__ _Float16 a1s[64 * 32];  // (attn*P1)^T tile: [m][n]
  __shared__ _Float16 a2s[64 * 32];
  __shared__ float diag[64];

  if (tid < 64) {
    const int m = m0 + tid;
    diag[tid] = attn[((size_t)b * N_NODES + m) * N_NODES + m];
  }

  const int msub = (wave & 3) * 16;        // which 16-row subtile
  const int colbase = (wave >> 2) * 192;   // which half of the 384 cols

  v8f acc[12] = {};

  // staging mapping: 32 rows (nn) x 8 groups of 8 consecutive m
  const int mg = tid & 7, nn = tid >> 3;
  const float* attn_b = attn + (size_t)b * N_NODES * N_NODES + m0 + mg * 8;
  const _Float16* Yb1 = Y + (size_t)(1 * B_SZ + b) * COLS * N_NODES;
  const _Float16* Yb2 = Y + (size_t)(2 * B_SZ + b) * COLS * N_NODES;

  for (int nb = 0; nb < N_NODES; nb += 32) {
    __syncthreads();
    {
      const size_t row = (size_t)(nb + nn);
      const float*    ar  = attn_b + row * N_NODES;
      const _Float16* p1r = P1 + row * N_NODES + m0 + mg * 8;
      const _Float16* p2r = P2 + row * N_NODES + m0 + mg * 8;
      const float4 av0 = *(const float4*)(ar);
      const float4 av1 = *(const float4*)(ar + 4);
      const v8h p1v = *(const v8h*)(p1r);
      const v8h p2v = *(const v8h*)(p2r);
      const float a[8] = {av0.x, av0.y, av0.z, av0.w, av1.x, av1.y, av1.z, av1.w};
      #pragma unroll
      for (int q = 0; q < 8; ++q) {
        const int mm = mg * 8 + q;
        a1s[mm * 32 + nn] = (_Float16)(a[q] * (float)p1v[q]);
        a2s[mm * 32 + nn] = (_Float16)(a[q] * (float)p2v[q]);
      }
    }
    __syncthreads();

    FragA A1, A2;
    const int abase = (msub + lm) * 32 + lhi * 8;
    A1.h[0] = *(const v8h*)&a1s[abase];
    A1.h[1] = *(const v8h*)&a1s[abase + 16];
    A2.h[0] = *(const v8h*)&a2s[abase];
    A2.h[1] = *(const v8h*)&a2s[abase + 16];

    #pragma unroll
    for (int ct = 0; ct < 12; ++ct) {
      const size_t coff = (size_t)(colbase + ct * 16 + lm) * N_NODES + nb + lhi * 8;
      FragA B1, B2;
      const _Float16* y1 = Yb1 + coff;
      const _Float16* y2 = Yb2 + coff;
      B1.h[0] = *(const v8h*)(y1);
      B1.h[1] = *(const v8h*)(y1 + 16);
      B2.h[0] = *(const v8h*)(y2);
      B2.h[1] = *(const v8h*)(y2 + 16);
      acc[ct] = __builtin_amdgcn_wmma_f32_16x16x32_f16(
          false, A1.v, false, B1.v, (short)0, acc[ct], false, false);
      acc[ct] = __builtin_amdgcn_wmma_f32_16x16x32_f16(
          false, A2.v, false, B2.v, (short)0, acc[ct], false, false);
    }
  }

  // epilogue: + diag(attn) * y0 (k=0 identity poly), relu, store fp32
  const _Float16* Yb0 = Y + (size_t)(0 * B_SZ + b) * COLS * N_NODES;
  #pragma unroll
  for (int ct = 0; ct < 12; ++ct) {
    const int col = colbase + ct * 16 + lm;
    const v8h y0v = *(const v8h*)(Yb0 + (size_t)col * N_NODES + m0 + msub + lhi * 8);
    #pragma unroll
    for (int r = 0; r < 8; ++r) {
      const int ml = msub + lhi * 8 + r;
      float val = acc[ct][r] + diag[ml] * (float)y0v[r];
      val = fmaxf(val, 0.0f);
      out[((size_t)b * N_NODES + m0 + ml) * COLS + col] = val;
    }
  }
}

// ---------------------------------------------------------------------------
extern "C" void kernel_launch(void* const* d_in, const int* in_sizes, int n_in,
                              void* d_out, int out_size, void* d_ws, size_t ws_size,
                              hipStream_t stream) {
  const float* x     = (const float*)d_in[0];  // (8,2048,16,12)
  const float* attn  = (const float*)d_in[1];  // (8,2048,2048)
  const float* E     = (const float*)d_in[2];  // (2048,16)
  const float* theta = (const float*)d_in[3];  // (3,16,32)
  float* out = (float*)d_out;                  // (8,2048,32,12)

  const size_t NN = (size_t)N_NODES * N_NODES;
  _Float16* S  = (_Float16*)d_ws;
  _Float16* St = S + NN;
  _Float16* T2 = St + NN;
  _Float16* Yw = T2 + NN;   // 3*8*384*2048 halves

  supports_kernel<<<N_NODES, 256, 0, stream>>>(E, S, St);
  cheb2_kernel<<<dim3(16, 32), 256, 0, stream>>>(S, St, T2);
  y_kernel<<<dim3(32, 8), 256, 0, stream>>>(x, theta, Yw);
  agcn_main_kernel<<<dim3(32, 8), 256, 0, stream>>>(attn, S, T2, Yw, out);
}
